// ChannelAttention_80685255623378
// MI455X (gfx1250) — matrix operand order repaired and checked by reading, not compiled, not run on hardware
//
#include <hip/hip_runtime.h>

#define CCH   64
#define NTOT  110592          // 48*48*48
#define NBATCH 8

typedef __attribute__((ext_vector_type(16))) __bf16 bf16x16;
typedef __attribute__((ext_vector_type(2)))  __bf16 bf16x2;
typedef __attribute__((ext_vector_type(8)))  float  f32x8;

union Frag { unsigned u[8]; bf16x16 v; };

// f32x2 -> packed bf16x2. gfx1250 clang does not declare cvt_pk_bf16_f32
// (probe-confirmed: fallback path was taken last round), so do it with one
// v_perm_b32: add 0x8000 (round-to-nearest, ties away) then gather the two
// high halves in a single byte-permute. 3 VALU ops per pair.
__device__ __forceinline__ unsigned pack_bf16x2(float lo, float hi) {
#if __has_builtin(__builtin_amdgcn_cvt_pk_bf16_f32)
    bf16x2 p = __builtin_amdgcn_cvt_pk_bf16_f32(lo, hi);
    return __builtin_bit_cast(unsigned, p);
#else
    unsigned a = __float_as_uint(lo) + 0x8000u;
    unsigned b = __float_as_uint(hi) + 0x8000u;
    // {s0,s1} byte pool: s1 = bytes 0..3, s0 = bytes 4..7.
    // dst = { b[3], b[2], a[3], a[2] }  ->  selector 0x07060302
    return __builtin_amdgcn_perm(b, a, 0x07060302u);
#endif
}

// ---------------------------------------------------------------- zero scratch
__global__ void zero_kernel(float* g, int n) {
    int i = blockIdx.x * blockDim.x + threadIdx.x;
    if (i < n) g[i] = 0.0f;
}

// ---------------------------------------------------------------- G = K K^T
// grid (54, 8), block 512 (16 waves). Each WG covers 2048 n-columns in 16
// tiles of 128, DOUBLE-BUFFERED in LDS: prefetch tile t+1 into registers,
// run 4 WMMAs on tile t, convert+store, one barrier per iteration.
// LDS tile: 64 ch x 128 n bf16, pitch 132 (66 words) -> b64-aligned rows,
// row bank-stride 2 avoids the 16-way conflict of pitch 128.
#define TILE_N   128
#define GPITCH_W 66
#define CHUNK_N  2048
#define GTILES   (CHUNK_N / TILE_N)

__global__ __launch_bounds__(512) void gram_kernel(const float* __restrict__ x,
                                                   float* __restrict__ G) {
    __shared__ unsigned lds[2][CCH * GPITCH_W];     // 2 x 16896 B

    const int b      = blockIdx.y;
    const int chunk0 = blockIdx.x * CHUNK_N;
    const int tid    = threadIdx.x;
    const int lane   = tid & 31;
    const int w      = tid >> 5;               // 0..15
    const int ci     = (w >> 2) * 16;
    const int cj     = (w & 3) * 16;
    const int mrow   = lane & 15;
    const int hi     = lane >> 4;              // 0 or 1

    const int lc = tid >> 5;                   // staging: channel sub-row
    const int ln = (tid & 31) * 4;             // staging: n offset (float4)

    const float* xb = x + (size_t)b * CCH * NTOT;

    // ---- prologue: stage tile 0 into buffer 0
    #pragma unroll
    for (int it = 0; it < 4; ++it) {
        const int c = lc + 16 * it;
        const float4 v4 = *(const float4*)(xb + (size_t)c * NTOT + chunk0 + ln);
        const int widx = c * GPITCH_W + (ln >> 1);
        lds[0][widx]     = pack_bf16x2(v4.x, v4.y);
        lds[0][widx + 1] = pack_bf16x2(v4.z, v4.w);
    }
    __syncthreads();

    f32x8 acc = {};

    for (int t = 0; t < GTILES; ++t) {
        const int cur = t & 1, nxt = cur ^ 1;
        const bool hasnext = (t + 1 < GTILES);

        // issue next tile's global loads early (latency overlaps the WMMAs)
        float4 pre[4];
        if (hasnext) {
            const int n1 = chunk0 + (t + 1) * TILE_N;
            #pragma unroll
            for (int it = 0; it < 4; ++it)
                pre[it] = *(const float4*)(xb + (size_t)(lc + 16 * it) * NTOT + n1 + ln);
        }

        // 4 k-chunks of 32 -> 4 WMMAs on current buffer
        #pragma unroll
        for (int k0 = 0; k0 < TILE_N; k0 += 32) {
            Frag A, B;
            #pragma unroll
            for (int v = 0; v < 8; ++v) {
                // 16-bit A 16x32 layout: lanes0-15 K={0..7,16..23}, lanes16-31 +8
                const int kkA = ((v < 4) ? 2 * v : 16 + 2 * (v - 4)) + hi * 8;
                A.u[v] = lds[cur][(ci + mrow) * GPITCH_W + ((k0 + kkA) >> 1)];
                // 16-bit B 32x16 layout: lanes0-15 K={0..7,16..23}, lanes16-31 +8
                const int kkB = ((v < 4) ? 2 * v : 16 + 2 * (v - 4)) + hi * 8;
                B.u[v] = lds[cur][(cj + mrow) * GPITCH_W + ((k0 + kkB) >> 1)];
            }
            acc = __builtin_amdgcn_wmma_f32_16x16x32_bf16(
                false, A.v, false, B.v, (short)0, acc, false, false);
        }

        // convert + store next tile, then one barrier covers both hazards
        if (hasnext) {
            #pragma unroll
            for (int it = 0; it < 4; ++it) {
                const int c = lc + 16 * it;
                const int widx = c * GPITCH_W + (ln >> 1);
                lds[nxt][widx]     = pack_bf16x2(pre[it].x, pre[it].y);
                lds[nxt][widx + 1] = pack_bf16x2(pre[it].z, pre[it].w);
            }
        }
        __syncthreads();
    }

    // D layout: VGPR v -> row v (lanes 0-15) / v+8 (lanes 16-31), col = lane&15
    float* Gb = G + b * CCH * CCH;
    const int ncol  = lane & 15;
    const int mbase = hi * 8;
    #pragma unroll
    for (int v = 0; v < 8; ++v)
        unsafeAtomicAdd(&Gb[(ci + mbase + v) * CCH + cj + ncol], acc[v]);
}

// ---------------------------------------------------------------- sigmoid(G*G)
// Tiny (4 MFLOP total): keep full fp32. One WG per batch.
__global__ __launch_bounds__(256) void affinity_kernel(const float* __restrict__ G,
                                                       float* __restrict__ Aff) {
    __shared__ float g[CCH * CCH];
    const int b = blockIdx.x;
    const float* Gb = G + b * CCH * CCH;
    for (int i = threadIdx.x; i < CCH * CCH; i += 256) g[i] = Gb[i];
    __syncthreads();
    const int c  = threadIdx.x >> 2;
    const int e0 = (threadIdx.x & 3) * 16;
    float* Ab = Aff + b * CCH * CCH;
    #pragma unroll 4
    for (int j = 0; j < 16; ++j) {
        const int e = e0 + j;
        float s = 0.0f;
        #pragma unroll 8
        for (int d = 0; d < CCH; ++d) s += g[c * CCH + d] * g[d * CCH + e];
        Ab[c * CCH + e] = 1.0f / (1.0f + __expf(-s));   // saturates cleanly
    }
}

// ---------------------------------------------------------------- out = g*(A@K)+x
// grid (432, 8), block 512. Each WG covers 256 n-columns (4 tiles of 64):
// affinity staged ONCE (bf16 row-major), its two A-fragments hoisted into
// registers (invariant across tiles); K tile staged TRANSPOSED [n][d] and
// double-buffered. Pitch 68 bf16 keeps rows b64-aligned, de-conflicts banks.
#define OPITCH_W 34
#define ONB      4            // 64-column tiles per WG

__global__ __launch_bounds__(512) void out_kernel(const float* __restrict__ x,
                                                  const float* __restrict__ Aff,
                                                  const float* __restrict__ gamma_p,
                                                  float* __restrict__ out) {
    __shared__ unsigned affl[CCH * OPITCH_W];       // 64 x 68 bf16
    __shared__ unsigned kt[2][CCH * OPITCH_W];      // [n][d] transposed tiles

    const int b    = blockIdx.y;
    const int nb0  = blockIdx.x * (64 * ONB);
    const int tid  = threadIdx.x;
    const int lane = tid & 31;
    const int w    = tid >> 5;
    const float gamma = gamma_p[0];

    const float* xb = x   + (size_t)b * CCH * NTOT;
    const float* Ab = Aff + b * CCH * CCH;
    float*       ob = out + (size_t)b * CCH * NTOT;

    const int sc = tid >> 4;                   // staging channel sub-row (0..31)
    const int sn = (tid & 15) * 4;             // staging n offset

    // ---- stage affinity f32 -> bf16 row-major (once per WG)
    #pragma unroll
    for (int it = 0; it < 2; ++it) {
        const int c = sc + 32 * it;
        const float4 v4 = *(const float4*)(Ab + c * CCH + sn);
        const int widx = c * OPITCH_W + (sn >> 1);
        affl[widx]     = pack_bf16x2(v4.x, v4.y);
        affl[widx + 1] = pack_bf16x2(v4.z, v4.w);
    }
    // ---- stage K tile 0 transposed [n][c] into buffer 0
    {
        unsigned short* kts = (unsigned short*)kt[0];
        #pragma unroll
        for (int it = 0; it < 2; ++it) {
            const int c = sc + 32 * it;
            const float4 v4 = *(const float4*)(xb + (size_t)c * NTOT + nb0 + sn);
            const unsigned p01 = pack_bf16x2(v4.x, v4.y);
            const unsigned p23 = pack_bf16x2(v4.z, v4.w);
            kts[(sn + 0) * (2 * OPITCH_W) + c] = (unsigned short)p01;
            kts[(sn + 1) * (2 * OPITCH_W) + c] = (unsigned short)(p01 >> 16);
            kts[(sn + 2) * (2 * OPITCH_W) + c] = (unsigned short)p23;
            kts[(sn + 3) * (2 * OPITCH_W) + c] = (unsigned short)(p23 >> 16);
        }
    }
    __syncthreads();

    const int ci   = (w >> 2) * 16;
    const int nj   = (w & 3) * 16;
    const int mrow = lane & 15;
    const int hi   = lane >> 4;

    // ---- hoist A fragments (affinity rows ci..ci+15; d-chunks 0 and 32)
    Frag A0, A1;
    #pragma unroll
    for (int v = 0; v < 8; ++v) {
        const int kkA = ((v < 4) ? 2 * v : 16 + 2 * (v - 4)) + hi * 8;
        A0.u[v] = affl[(ci + mrow) * OPITCH_W + ((     kkA) >> 1)];
        A1.u[v] = affl[(ci + mrow) * OPITCH_W + ((32 + kkA) >> 1)];
    }

    for (int t = 0; t < ONB; ++t) {
        const int cur = t & 1, nxt = cur ^ 1;
        const bool hasnext = (t + 1 < ONB);
        const int nb = nb0 + t * 64;

        // prefetch next tile into registers
        float4 pre[2];
        if (hasnext) {
            const int nb1 = nb + 64;
            #pragma unroll
            for (int it = 0; it < 2; ++it)
                pre[it] = *(const float4*)(xb + (size_t)(sc + 32 * it) * NTOT + nb1 + sn);
        }

        // ---- 2 WMMAs: acc = Aff[ci,:] @ K[:, nb+nj..+15]
        f32x8 acc = {};
        #pragma unroll
        for (int half = 0; half < 2; ++half) {
            const int d0 = half * 32;
            Frag B;
            #pragma unroll
            for (int v = 0; v < 8; ++v) {
                const int kkB = ((v < 4) ? 2 * v : 16 + 2 * (v - 4)) + hi * 8;
                B.u[v] = kt[cur][(nj + mrow) * OPITCH_W + ((d0 + kkB) >> 1)];
            }
            acc = __builtin_amdgcn_wmma_f32_16x16x32_bf16(
                false, (half ? A1.v : A0.v), false, B.v, (short)0, acc, false, false);
        }

        // ---- out[c][n] = gamma*acc + x[c][n]  (fp32 residual, coalesced rows)
        const int ncol  = nb + nj + (lane & 15);
        const int mbase = ci + hi * 8;
        #pragma unroll
        for (int v = 0; v < 8; ++v) {
            const size_t idx = (size_t)(mbase + v) * NTOT + ncol;
            ob[idx] = fmaf(gamma, acc[v], xb[idx]);
        }

        // ---- convert + store next tile, single barrier per iteration
        if (hasnext) {
            unsigned short* kts = (unsigned short*)kt[nxt];
            #pragma unroll
            for (int it = 0; it < 2; ++it) {
                const int c = sc + 32 * it;
                const unsigned p01 = pack_bf16x2(pre[it].x, pre[it].y);
                const unsigned p23 = pack_bf16x2(pre[it].z, pre[it].w);
                kts[(sn + 0) * (2 * OPITCH_W) + c] = (unsigned short)p01;
                kts[(sn + 1) * (2 * OPITCH_W) + c] = (unsigned short)(p01 >> 16);
                kts[(sn + 2) * (2 * OPITCH_W) + c] = (unsigned short)p23;
                kts[(sn + 3) * (2 * OPITCH_W) + c] = (unsigned short)(p23 >> 16);
            }
        }
        __syncthreads();
    }
}

// ---------------------------------------------------------------- launch
extern "C" void kernel_launch(void* const* d_in, const int* in_sizes, int n_in,
                              void* d_out, int out_size, void* d_ws, size_t ws_size,
                              hipStream_t stream) {
    const float* x     = (const float*)d_in[0];
    const float* gamma = (const float*)d_in[1];
    float* out = (float*)d_out;

    float* G   = (float*)d_ws;                    // 8*64*64 f32
    float* Aff = G + NBATCH * CCH * CCH;          // 8*64*64 f32

    const int gelems = NBATCH * CCH * CCH;
    hipLaunchKernelGGL(zero_kernel, dim3((gelems + 255) / 256), dim3(256), 0, stream,
                       G, gelems);
    hipLaunchKernelGGL(gram_kernel, dim3(NTOT / CHUNK_N, NBATCH), dim3(512), 0, stream,
                       x, G);
    hipLaunchKernelGGL(affinity_kernel, dim3(NBATCH), dim3(256), 0, stream, G, Aff);
    hipLaunchKernelGGL(out_kernel, dim3(NTOT / (64 * ONB), NBATCH), dim3(512), 0, stream,
                       x, Aff, gamma, out);
}
